// RWKV7_CustomCrossAttnBlock_78391743086614
// MI455X (gfx1250) — matrix operand order repaired and checked by reading, not compiled, not run on hardware
//
#include <hip/hip_runtime.h>
#include <hip/hip_bf16.h>

// ---------------------------------------------------------------------------
// RWKV7 cross-attn block for MI455X (gfx1250, wave32, WMMA).
// All matmuls go through v_wmma_f32_16x16x32_bf16; sequential WKV scan keeps
// the 64x64 state entirely in VGPRs (one wave per (b,h)).
// ---------------------------------------------------------------------------

#define B_     4
#define TQ_    256
#define TKV_   1024
#define C_     1024
#define H_     16
#define D_     64
#define FF_    4096
#define W_SCALE_ (-0.6065306597126334f)

typedef __attribute__((ext_vector_type(16))) __bf16 v16bf;
typedef __attribute__((ext_vector_type(8)))  float  v8f;

__device__ __forceinline__ unsigned short f2bf(float f) {
  unsigned int u = __float_as_uint(f);
  unsigned int r = u + 0x7FFFu + ((u >> 16) & 1u);   // round-to-nearest-even
  return (unsigned short)(r >> 16);
}
__device__ __forceinline__ float sigmoidf_(float x) { return 1.0f / (1.0f + __expf(-x)); }
__device__ __forceinline__ float wave_sum32(float x) {
  #pragma unroll
  for (int m = 16; m > 0; m >>= 1) x += __shfl_xor(x, m, 32);
  return x;
}

// ---------------------------------------------------------------------------
// WMMA GEMM: C[M,N] (f32) = A[M,K] (bf16, row-major) * B[K,N] (bf16, row-major)
// Block tile 128x64, 8 waves (4x2), each wave 32x32 (2x2 WMMA tiles), K step 32.
// Requires M%128==0, N%64==0, K%32==0 (true for every GEMM in this block).
// ---------------------------------------------------------------------------
#define LDSPAD 36

union FragAB { v16bf v; unsigned int u[8]; };

__global__ __launch_bounds__(256)
void gemm_bf16_wmma(const unsigned short* __restrict__ A,
                    const unsigned short* __restrict__ Bw,
                    float* __restrict__ Co,
                    int M, int N, int K)
{
  __shared__ unsigned short ldsA[128 * LDSPAD];   // row-major [128][36]
  __shared__ unsigned short ldsB[64 * LDSPAD];    // transposed [N=64][K=36]

  const int tid  = threadIdx.x;
  const int lane = tid & 31, wid = tid >> 5;
  const int wm = wid & 3, wn = wid >> 2;
  const int half = lane >> 4, l16 = lane & 15;
  const int m0 = blockIdx.y * 128, n0 = blockIdx.x * 64;

  v8f acc[2][2];
  #pragma unroll
  for (int i = 0; i < 2; ++i)
    #pragma unroll
    for (int j = 0; j < 2; ++j)
      #pragma unroll
      for (int e = 0; e < 8; ++e) acc[i][j][e] = 0.0f;

  for (int k0 = 0; k0 < K; k0 += 32) {
    // --- stage A tile: 128x32 bf16, 16 elems/thread ---
    #pragma unroll
    for (int i = 0; i < 2; ++i) {
      int c = tid + i * 256;
      int row = c >> 2, col = (c & 3) * 8;
      uint4 d = *(const uint4*)(A + (size_t)(m0 + row) * K + k0 + col);
      const unsigned short* s = (const unsigned short*)&d;
      #pragma unroll
      for (int j = 0; j < 8; ++j) ldsA[row * LDSPAD + col + j] = s[j];
    }
    // --- stage B tile: 32x64 bf16, stored transposed ---
    {
      int kk = tid >> 3, nb = (tid & 7) * 8;
      uint4 d = *(const uint4*)(Bw + (size_t)(k0 + kk) * N + n0 + nb);
      const unsigned short* s = (const unsigned short*)&d;
      #pragma unroll
      for (int j = 0; j < 8; ++j) ldsB[(nb + j) * LDSPAD + kk] = s[j];
    }
    // prefetch next K tile (global_prefetch_b8)
    if (k0 + 32 < K) {
      __builtin_prefetch(A + (size_t)(m0 + (tid >> 1)) * K + k0 + 32, 0, 0);
      __builtin_prefetch(Bw + (size_t)(k0 + 32 + (tid >> 3)) * N + n0, 0, 0);
    }
    __syncthreads();

    FragAB fa[2], fb[2];
    #pragma unroll
    for (int mt = 0; mt < 2; ++mt) {
      int row = wm * 32 + mt * 16 + l16;
      #pragma unroll
      for (int p = 0; p < 8; ++p) {
        // A layout: VGPR p holds K pair base {0,2,4,6,16,18,20,22} + 8*half
        int kp = ((p < 4) ? (p * 2) : (16 + (p - 4) * 2)) + half * 8;
        fa[mt].u[p] = *(const unsigned int*)&ldsA[row * LDSPAD + kp];
      }
    }
    #pragma unroll
    for (int nt = 0; nt < 2; ++nt) {
      int colb = wn * 32 + nt * 16 + l16;
      #pragma unroll
      for (int p = 0; p < 8; ++p) {
        // B layout: VGPR p holds K pair base {0,2,4,6,16,18,20,22} + 8*half
        int kp = ((p < 4) ? (p * 2) : (16 + (p - 4) * 2)) + half * 8;
        fb[nt].u[p] = *(const unsigned int*)&ldsB[colb * LDSPAD + kp];
      }
    }
    #pragma unroll
    for (int mt = 0; mt < 2; ++mt)
      #pragma unroll
      for (int nt = 0; nt < 2; ++nt)
        acc[mt][nt] = __builtin_amdgcn_wmma_f32_16x16x32_bf16(
            false, fa[mt].v, false, fb[nt].v, (short)0, acc[mt][nt], false, false);
    __syncthreads();
  }

  // D layout: lane l: n = l%16, m = vgpr + 8*(l/16)
  #pragma unroll
  for (int mt = 0; mt < 2; ++mt)
    #pragma unroll
    for (int nt = 0; nt < 2; ++nt)
      #pragma unroll
      for (int j = 0; j < 8; ++j) {
        int m = m0 + wm * 32 + mt * 16 + half * 8 + j;
        int n = n0 + wn * 32 + nt * 16 + l16;
        Co[(size_t)m * N + n] = acc[mt][nt][j];
      }
}

// ---------------------------------------------------------------------------
// LayerNorm over last dim (C=1024), one block (256 thr) per row, f32 out.
// ---------------------------------------------------------------------------
__global__ __launch_bounds__(256)
void ln_rows(const float* __restrict__ x, const float* __restrict__ w,
             const float* __restrict__ b, float* __restrict__ y, int C)
{
  const int row = blockIdx.x, tid = threadIdx.x;
  const float* xr = x + (size_t)row * C;
  __shared__ float red[256];
  float s = 0.f;
  for (int c = tid; c < C; c += 256) s += xr[c];
  red[tid] = s; __syncthreads();
  for (int off = 128; off > 0; off >>= 1) { if (tid < off) red[tid] += red[tid + off]; __syncthreads(); }
  float mu = red[0] / C; __syncthreads();
  float v = 0.f;
  for (int c = tid; c < C; c += 256) { float d = xr[c] - mu; v += d * d; }
  red[tid] = v; __syncthreads();
  for (int off = 128; off > 0; off >>= 1) { if (tid < off) red[tid] += red[tid + off]; __syncthreads(); }
  float rstd = rsqrtf(red[0] / C + 1e-5f);
  float* yr = y + (size_t)row * C;
  for (int c = tid; c < C; c += 256) yr[c] = (xr[c] - mu) * rstd * w[c] + b[c];
}

// ---------------------------------------------------------------------------
// Token-shift mix -> bf16.  shsrc==nullptr: sequence shift (zero at t==0);
// else shifted[b] = shsrc[b*sh_stride + c] (frozen conv token).
// ---------------------------------------------------------------------------
__global__ __launch_bounds__(256)
void mix_to_bf16(const float* __restrict__ x, const float* __restrict__ xmix, int j,
                 const float* __restrict__ shsrc, long sh_stride,
                 unsigned short* __restrict__ out, int T, int C, long total)
{
  for (long idx = (long)blockIdx.x * 256 + threadIdx.x; idx < total; idx += (long)gridDim.x * 256) {
    long row = idx / C; int c = (int)(idx % C);
    long bb = row / T;  int t = (int)(row % T);
    float xv = x[idx];
    float sh = shsrc ? shsrc[bb * sh_stride + c] : ((t > 0) ? x[idx - C] : 0.f);
    out[idx] = f2bf(xv + (sh - xv) * xmix[(size_t)j * C + c]);
  }
}

// ---- elementwise helpers ---------------------------------------------------
__global__ void k_cvt_bf16(const float* __restrict__ x, unsigned short* __restrict__ y, long n) {
  for (long i = (long)blockIdx.x * 256 + threadIdx.x; i < n; i += (long)gridDim.x * 256) y[i] = f2bf(x[i]);
}
__global__ void k_tanh_bf16(const float* __restrict__ x, unsigned short* __restrict__ y, long n) {
  for (long i = (long)blockIdx.x * 256 + threadIdx.x; i < n; i += (long)gridDim.x * 256) y[i] = f2bf(tanhf(x[i]));
}
__global__ void k_sigmoid_bf16(const float* __restrict__ x, unsigned short* __restrict__ y, long n) {
  for (long i = (long)blockIdx.x * 256 + threadIdx.x; i < n; i += (long)gridDim.x * 256) y[i] = f2bf(sigmoidf_(x[i]));
}
__global__ void k_decay(const float* __restrict__ x, const float* __restrict__ w0,
                        float* __restrict__ y, long n, int C) {
  for (long i = (long)blockIdx.x * 256 + threadIdx.x; i < n; i += (long)gridDim.x * 256)
    y[i] = __expf(W_SCALE_ * sigmoidf_(x[i] + w0[i % C]));
}
__global__ void k_sigmoid_bias(const float* __restrict__ x, const float* __restrict__ b0,
                               float* __restrict__ y, long n, int C) {
  for (long i = (long)blockIdx.x * 256 + threadIdx.x; i < n; i += (long)gridDim.x * 256)
    y[i] = sigmoidf_(x[i] + b0[i % C]);
}
__global__ void k_sqrelu_bf16(const float* __restrict__ x, unsigned short* __restrict__ y, long n) {
  for (long i = (long)blockIdx.x * 256 + threadIdx.x; i < n; i += (long)gridDim.x * 256) {
    float r = fmaxf(x[i], 0.f); y[i] = f2bf(r * r);
  }
}
__global__ void k_add(const float* __restrict__ a, const float* __restrict__ b,
                      float* __restrict__ y, long n) {
  for (long i = (long)blockIdx.x * 256 + threadIdx.x; i < n; i += (long)gridDim.x * 256) y[i] = a[i] + b[i];
}

// ---------------------------------------------------------------------------
// Per-head prep: kk = l2norm(k*k_k); av=-kk; bv=kk*a; k <- k*(1+(a-1)*k_a).
// One wave per (row, head); lane owns d and d+32.
// ---------------------------------------------------------------------------
__global__ __launch_bounds__(256)
void prep_heads(float* __restrict__ k, const float* __restrict__ a,
                const float* __restrict__ k_k, const float* __restrict__ k_a,
                float* __restrict__ av, float* __restrict__ bv, long rows)
{
  long wave = (long)blockIdx.x * 8 + (threadIdx.x >> 5);
  if (wave >= rows * H_) return;
  int lane = threadIdx.x & 31;
  long row = wave >> 4; int h = (int)(wave & 15);
  long base = row * C_ + h * D_;
  int c0 = h * D_ + lane, c1 = c0 + 32;
  float k0v = k[base + lane], k1v = k[base + lane + 32];
  float kk0 = k0v * k_k[c0], kk1 = k1v * k_k[c1];
  float s = wave_sum32(kk0 * kk0 + kk1 * kk1);
  float inv = rsqrtf(s + 1e-12f);
  float a0 = a[base + lane], a1 = a[base + lane + 32];
  float n0 = kk0 * inv, n1 = kk1 * inv;
  av[base + lane]      = -n0;       av[base + lane + 32] = -n1;
  bv[base + lane]      =  n0 * a0;  bv[base + lane + 32] =  n1 * a1;
  k[base + lane]       = k0v * (1.f + (a0 - 1.f) * k_a[c0]);
  k[base + lane + 32]  = k1v * (1.f + (a1 - 1.f) * k_a[c1]);
}

// ---------------------------------------------------------------------------
// WKV scan: one wave per (b,h), state S[64x64] in VGPRs (2 v-rows / lane).
// S = S*diag(w) + (S av) bv^T + v k^T   (sa from pre-update S)
// ---------------------------------------------------------------------------
__global__ __launch_bounds__(32)
void wkv_scan(const float* __restrict__ w, const float* __restrict__ k,
              const float* __restrict__ v, const float* __restrict__ av,
              const float* __restrict__ bv, float* __restrict__ Sout, int T)
{
  int bh = blockIdx.x;              // b*H + h
  int b = bh >> 4, h = bh & 15;
  int lane = threadIdx.x;
  __shared__ alignas(16) float lw[64], lk[64], la[64], lb[64];
  float S0[64], S1[64];
  #pragma unroll
  for (int i = 0; i < 64; ++i) { S0[i] = 0.f; S1[i] = 0.f; }

  for (int t = 0; t < T; ++t) {
    long base = ((long)(b * T + t)) * C_ + h * D_;
    lw[lane] = w[base + lane];  lw[lane + 32] = w[base + lane + 32];
    lk[lane] = k[base + lane];  lk[lane + 32] = k[base + lane + 32];
    la[lane] = av[base + lane]; la[lane + 32] = av[base + lane + 32];
    lb[lane] = bv[base + lane]; lb[lane + 32] = bv[base + lane + 32];
    float vt0 = v[base + lane], vt1 = v[base + lane + 32];
    __syncthreads();

    float sa0 = 0.f, sa1 = 0.f;
    const float4* a4 = (const float4*)la;
    #pragma unroll
    for (int q = 0; q < 16; ++q) {
      float4 aq = a4[q];
      sa0 += S0[4*q+0]*aq.x + S0[4*q+1]*aq.y + S0[4*q+2]*aq.z + S0[4*q+3]*aq.w;
      sa1 += S1[4*q+0]*aq.x + S1[4*q+1]*aq.y + S1[4*q+2]*aq.z + S1[4*q+3]*aq.w;
    }
    const float4* w4 = (const float4*)lw;
    const float4* k4 = (const float4*)lk;
    const float4* b4 = (const float4*)lb;
    #pragma unroll
    for (int q = 0; q < 16; ++q) {
      float4 wq = w4[q], kq = k4[q], bq = b4[q];
      S0[4*q+0] = S0[4*q+0]*wq.x + sa0*bq.x + vt0*kq.x;
      S0[4*q+1] = S0[4*q+1]*wq.y + sa0*bq.y + vt0*kq.y;
      S0[4*q+2] = S0[4*q+2]*wq.z + sa0*bq.z + vt0*kq.z;
      S0[4*q+3] = S0[4*q+3]*wq.w + sa0*bq.w + vt0*kq.w;
      S1[4*q+0] = S1[4*q+0]*wq.x + sa1*bq.x + vt1*kq.x;
      S1[4*q+1] = S1[4*q+1]*wq.y + sa1*bq.y + vt1*kq.y;
      S1[4*q+2] = S1[4*q+2]*wq.z + sa1*bq.z + vt1*kq.z;
      S1[4*q+3] = S1[4*q+3]*wq.w + sa1*bq.w + vt1*kq.w;
    }
    __syncthreads();
  }
  float* out = Sout + (long)bh * D_ * D_;
  #pragma unroll
  for (int kk2 = 0; kk2 < 64; ++kk2) {
    out[(long)lane * 64 + kk2]        = S0[kk2];
    out[(long)(lane + 32) * 64 + kk2] = S1[kk2];
  }
}

// ---------------------------------------------------------------------------
// Query attention: one step from frozen S per (b,t,h). Wave-per-task.
// o = S.(w*r) + (S.av)*(bv.r) + v*(k.r)
// ---------------------------------------------------------------------------
__global__ __launch_bounds__(256)
void query_attn_core(const float* __restrict__ S, const float* __restrict__ r,
                     const float* __restrict__ w, const float* __restrict__ k,
                     const float* __restrict__ av, const float* __restrict__ bv,
                     const float* __restrict__ v, float* __restrict__ oheads, int TQ)
{
  __shared__ alignas(16) float ls[8][5][64];   // per wave: r,w,k,av,bv
  int wid = threadIdx.x >> 5, lane = threadIdx.x & 31;
  long task = (long)blockIdx.x * 8 + wid;      // (b*TQ+t)*H + h
  long row = task >> 4; int h = (int)(task & 15);
  long base = row * C_ + h * D_;
  float (*L)[64] = ls[wid];
  L[0][lane] = r[base + lane];  L[0][lane + 32] = r[base + lane + 32];
  L[1][lane] = w[base + lane];  L[1][lane + 32] = w[base + lane + 32];
  L[2][lane] = k[base + lane];  L[2][lane + 32] = k[base + lane + 32];
  L[3][lane] = av[base + lane]; L[3][lane + 32] = av[base + lane + 32];
  L[4][lane] = bv[base + lane]; L[4][lane + 32] = bv[base + lane + 32];
  __syncthreads();

  float dbr = 0.f, dkr = 0.f;
  #pragma unroll 8
  for (int kk2 = 0; kk2 < 64; ++kk2) {
    float rr = L[0][kk2]; dbr += L[4][kk2] * rr; dkr += L[2][kk2] * rr;
  }
  long b = row / TQ;
  const float* Sr0 = S + (((long)b * H_ + h) * D_ + lane) * D_;
  const float* Sr1 = S + (((long)b * H_ + h) * D_ + lane + 32) * D_;
  float t10 = 0.f, t11 = 0.f, sa0 = 0.f, sa1 = 0.f;
  #pragma unroll 8
  for (int kk2 = 0; kk2 < 64; ++kk2) {
    float wr = L[1][kk2] * L[0][kk2], aa = L[3][kk2];
    float s0 = Sr0[kk2], s1 = Sr1[kk2];
    t10 += s0 * wr; sa0 += s0 * aa;
    t11 += s1 * wr; sa1 += s1 * aa;
  }
  float vt0 = v[base + lane], vt1 = v[base + lane + 32];
  oheads[base + lane]      = t10 + sa0 * dbr + vt0 * dkr;
  oheads[base + lane + 32] = t11 + sa1 * dbr + vt1 * dkr;
}

// ---------------------------------------------------------------------------
// Finish: per-head group norm (eps = D*1e-5) + rwkv bonus + gate -> bf16.
// ---------------------------------------------------------------------------
__global__ __launch_bounds__(256)
void finish_gate(const float* __restrict__ oheads, const float* __restrict__ r,
                 const float* __restrict__ k, const float* __restrict__ v,
                 const float* __restrict__ g, const float* __restrict__ r_k,
                 const float* __restrict__ gn_w, const float* __restrict__ gn_b,
                 unsigned short* __restrict__ preout)
{
  int wid = threadIdx.x >> 5, lane = threadIdx.x & 31;
  long task = (long)blockIdx.x * 8 + wid;
  long row = task >> 4; int h = (int)(task & 15);
  long base = row * C_ + h * D_;
  float o0 = oheads[base + lane], o1 = oheads[base + lane + 32];
  float mu = wave_sum32(o0 + o1) * (1.f / 64.f);
  float d0 = o0 - mu, d1 = o1 - mu;
  float var = wave_sum32(d0 * d0 + d1 * d1) * (1.f / 64.f);
  float rstd = rsqrtf(var + 64.0f * 1e-5f);
  int c0 = h * D_ + lane, c1 = c0 + 32;
  float bon = r[base + lane] * k[base + lane] * r_k[c0]
            + r[base + lane + 32] * k[base + lane + 32] * r_k[c1];
  float dot = wave_sum32(bon);
  float p0 = ((d0 * rstd) * gn_w[c0] + gn_b[c0] + dot * v[base + lane])      * g[base + lane];
  float p1 = ((d1 * rstd) * gn_w[c1] + gn_b[c1] + dot * v[base + lane + 32]) * g[base + lane + 32];
  preout[base + lane]      = f2bf(p0);
  preout[base + lane + 32] = f2bf(p1);
}

// ---------------------------------------------------------------------------
// FFN token-shift mix -> bf16 (sequence shift within query sequence).
// ---------------------------------------------------------------------------
__global__ __launch_bounds__(256)
void ffn_mix(const float* __restrict__ hbuf, const float* __restrict__ xk,
             unsigned short* __restrict__ out, int T, int C, long total)
{
  for (long idx = (long)blockIdx.x * 256 + threadIdx.x; idx < total; idx += (long)gridDim.x * 256) {
    long row = idx / C; int c = (int)(idx % C); int t = (int)(row % T);
    float hv = hbuf[idx];
    float sh = (t > 0) ? hbuf[idx - C] : 0.f;
    out[idx] = f2bf(hv + (sh - hv) * xk[c]);
  }
}

// ---------------------------------------------------------------------------
// Host orchestration
// ---------------------------------------------------------------------------
extern "C" void kernel_launch(void* const* d_in, const int* in_sizes, int n_in,
                              void* d_out, int out_size, void* d_ws, size_t ws_size,
                              hipStream_t stream)
{
  (void)in_sizes; (void)n_in; (void)out_size; (void)ws_size;
  const long R_KV = (long)B_ * TKV_;   // 4096
  const long R_Q  = (long)B_ * TQ_;    // 1024

  // -- input map (setup_inputs dict order) --
  const float* query  = (const float*)d_in[0];
  const float* keyval = (const float*)d_in[1];
  const int QP = 2, KP = 20, BP = 38;
  const float* q_xmix = (const float*)d_in[QP+0];
  const float* q_Wr   = (const float*)d_in[QP+1];
  const float* q_Wk   = (const float*)d_in[QP+2];
  const float* q_Wv   = (const float*)d_in[QP+3];
  const float* q_Wo   = (const float*)d_in[QP+4];
  const float* q_w1   = (const float*)d_in[QP+5];
  const float* q_w2   = (const float*)d_in[QP+6];
  const float* q_w0   = (const float*)d_in[QP+7];
  const float* q_a1   = (const float*)d_in[QP+8];
  const float* q_a2   = (const float*)d_in[QP+9];
  const float* q_a0   = (const float*)d_in[QP+10];
  const float* q_g1   = (const float*)d_in[QP+11];
  const float* q_g2   = (const float*)d_in[QP+12];
  const float* q_kk   = (const float*)d_in[QP+13];
  const float* q_ka   = (const float*)d_in[QP+14];
  const float* q_rk   = (const float*)d_in[QP+15];
  const float* q_gnw  = (const float*)d_in[QP+16];
  const float* q_gnb  = (const float*)d_in[QP+17];
  const float* k_xmix = (const float*)d_in[KP+0];
  const float* k_Wk   = (const float*)d_in[KP+2];
  const float* k_Wv   = (const float*)d_in[KP+3];
  const float* k_w1   = (const float*)d_in[KP+5];
  const float* k_w2   = (const float*)d_in[KP+6];
  const float* k_w0   = (const float*)d_in[KP+7];
  const float* k_a1   = (const float*)d_in[KP+8];
  const float* k_a2   = (const float*)d_in[KP+9];
  const float* k_a0   = (const float*)d_in[KP+10];
  const float* k_kkp  = (const float*)d_in[KP+13];
  const float* k_kap  = (const float*)d_in[KP+14];
  const float* pre_w  = (const float*)d_in[BP+0];
  const float* pre_b  = (const float*)d_in[BP+1];
  const float* qn_w   = (const float*)d_in[BP+2];
  const float* qn_b   = (const float*)d_in[BP+3];
  const float* kvn_w  = (const float*)d_in[BP+4];
  const float* kvn_b  = (const float*)d_in[BP+5];
  const float* ffnn_w = (const float*)d_in[BP+6];
  const float* ffnn_b = (const float*)d_in[BP+7];
  const float* ffn_xk = (const float*)d_in[BP+8];
  const float* ffn_Wk = (const float*)d_in[BP+9];
  const float* ffn_Wv = (const float*)d_in[BP+10];

  float* outF   = (float*)d_out;                 // [R_Q*C] residual+ffn
  float* v_q    = outF + R_Q * C_;               // [R_Q*C] query_v_first
  float* v_kv   = v_q + R_Q * C_;                // [R_KV*C] kv_v_first

  // -- workspace bump allocator --
  char* wsp = (char*)d_ws; size_t off = 0;
  auto alloc = [&](size_t bytes) -> void* {
    void* p = wsp + off; off += (bytes + 255) & ~(size_t)255; return p;
  };
  auto fbuf = [&](long n) { return (float*)alloc((size_t)n * 4); };
  auto hbuf_ = [&](long n) { return (unsigned short*)alloc((size_t)n * 2); };

  // bf16 weights
  unsigned short *kWk_b = hbuf_(C_*C_), *kWv_b = hbuf_(C_*C_);
  unsigned short *kw1_b = hbuf_(C_*64), *kw2_b = hbuf_(64*C_);
  unsigned short *ka1_b = hbuf_(C_*64), *ka2_b = hbuf_(64*C_);
  unsigned short *qWr_b = hbuf_(C_*C_), *qWk_b = hbuf_(C_*C_);
  unsigned short *qWv_b = hbuf_(C_*C_), *qWo_b = hbuf_(C_*C_);
  unsigned short *qw1_b = hbuf_(C_*64), *qw2_b = hbuf_(64*C_);
  unsigned short *qa1_b = hbuf_(C_*64), *qa2_b = hbuf_(64*C_);
  unsigned short *qg1_b = hbuf_(C_*128), *qg2_b = hbuf_(128*C_);
  unsigned short *fWk_b = hbuf_((long)C_*FF_), *fWv_b = hbuf_((long)FF_*C_);
  // activations
  float *kvx = fbuf(R_KV*C_);
  unsigned short *mix_kv = hbuf_(R_KV*C_);
  float *tmp64 = fbuf(R_KV*64); unsigned short *tmp64b = hbuf_(R_KV*64);
  float *tmpC_kv = fbuf(R_KV*C_);
  float *w_kv = fbuf(R_KV*C_), *k_kv = fbuf(R_KV*C_), *a_kv = fbuf(R_KV*C_);
  float *av_kv = fbuf(R_KV*C_), *bv_kv = fbuf(R_KV*C_);
  float *Sst = fbuf((long)B_*H_*D_*D_);
  float *resid0 = fbuf(R_Q*C_), *qx = fbuf(R_Q*C_);
  unsigned short *mix_q = hbuf_(R_Q*C_);
  float *tmpC_q = fbuf(R_Q*C_);
  float *r_q = fbuf(R_Q*C_), *w_q = fbuf(R_Q*C_), *k_q = fbuf(R_Q*C_);
  float *a_q = fbuf(R_Q*C_), *g_q = fbuf(R_Q*C_);
  float *av_q = fbuf(R_Q*C_), *bv_q = fbuf(R_Q*C_);
  float *tmp128 = fbuf(R_Q*128); unsigned short *tmp128b = hbuf_(R_Q*128);
  float *ohd = fbuf(R_Q*C_); unsigned short *preout = hbuf_(R_Q*C_);
  float *attn = fbuf(R_Q*C_);
  float *hln = fbuf(R_Q*C_); unsigned short *keyin = hbuf_(R_Q*C_);
  float *ffnmid = fbuf(R_Q*(long)FF_); unsigned short *ffnmid_b = hbuf_(R_Q*(long)FF_);
  float *ffnout = fbuf(R_Q*C_);

  auto GS = [](long n) { return (unsigned)((n + 255) / 256); };
  auto cvt = [&](const float* s, unsigned short* d, long n) {
    k_cvt_bf16<<<GS(n), 256, 0, stream>>>(s, d, n);
  };
  auto gemm = [&](const unsigned short* A, const unsigned short* Bw, float* Cm,
                  int M, int N, int K) {
    dim3 g((unsigned)(N / 64), (unsigned)(M / 128));
    gemm_bf16_wmma<<<g, 256, 0, stream>>>(A, Bw, Cm, M, N, K);
  };

  // ============================ KV PATH ============================
  ln_rows<<<(unsigned)R_KV, 256, 0, stream>>>(keyval, kvn_w, kvn_b, kvx, C_);
  cvt(k_Wk, kWk_b, (long)C_*C_);  cvt(k_Wv, kWv_b, (long)C_*C_);
  cvt(k_w1, kw1_b, (long)C_*64);  cvt(k_w2, kw2_b, (long)64*C_);
  cvt(k_a1, ka1_b, (long)C_*64);  cvt(k_a2, ka2_b, (long)64*C_);

  // decay lora: w = exp(W_SCALE*sigmoid(tanh(xw@w1)@w2 + w0))
  mix_to_bf16<<<GS(R_KV*C_), 256, 0, stream>>>(kvx, k_xmix, 1, nullptr, 0, mix_kv, TKV_, C_, R_KV*C_);
  gemm(mix_kv, kw1_b, tmp64, (int)R_KV, 64, C_);
  k_tanh_bf16<<<GS(R_KV*64), 256, 0, stream>>>(tmp64, tmp64b, R_KV*64);
  gemm(tmp64b, kw2_b, tmpC_kv, (int)R_KV, C_, 64);
  k_decay<<<GS(R_KV*C_), 256, 0, stream>>>(tmpC_kv, k_w0, w_kv, R_KV*C_, C_);
  // k
  mix_to_bf16<<<GS(R_KV*C_), 256, 0, stream>>>(kvx, k_xmix, 2, nullptr, 0, mix_kv, TKV_, C_, R_KV*C_);
  gemm(mix_kv, kWk_b, k_kv, (int)R_KV, C_, C_);
  // v  (kv_v_first -> straight into d_out region)
  mix_to_bf16<<<GS(R_KV*C_), 256, 0, stream>>>(kvx, k_xmix, 3, nullptr, 0, mix_kv, TKV_, C_, R_KV*C_);
  gemm(mix_kv, kWv_b, v_kv, (int)R_KV, C_, C_);
  // a = sigmoid(xa@a1@a2 + a0)
  mix_to_bf16<<<GS(R_KV*C_), 256, 0, stream>>>(kvx, k_xmix, 4, nullptr, 0, mix_kv, TKV_, C_, R_KV*C_);
  gemm(mix_kv, ka1_b, tmp64, (int)R_KV, 64, C_);
  k_cvt_bf16<<<GS(R_KV*64), 256, 0, stream>>>(tmp64, tmp64b, R_KV*64);
  gemm(tmp64b, ka2_b, tmpC_kv, (int)R_KV, C_, 64);
  k_sigmoid_bias<<<GS(R_KV*C_), 256, 0, stream>>>(tmpC_kv, k_a0, a_kv, R_KV*C_, C_);

  prep_heads<<<(unsigned)(R_KV*H_/8), 256, 0, stream>>>(k_kv, a_kv, k_kkp, k_kap, av_kv, bv_kv, R_KV);
  wkv_scan<<<(unsigned)(B_*H_), 32, 0, stream>>>(w_kv, k_kv, v_kv, av_kv, bv_kv, Sst, TKV_);

  // ============================ QUERY PATH ============================
  ln_rows<<<(unsigned)R_Q, 256, 0, stream>>>(query,  pre_w, pre_b, resid0, C_);
  ln_rows<<<(unsigned)R_Q, 256, 0, stream>>>(resid0, qn_w,  qn_b,  qx,     C_);
  cvt(q_Wr, qWr_b, (long)C_*C_); cvt(q_Wk, qWk_b, (long)C_*C_);
  cvt(q_Wv, qWv_b, (long)C_*C_); cvt(q_Wo, qWo_b, (long)C_*C_);
  cvt(q_w1, qw1_b, (long)C_*64); cvt(q_w2, qw2_b, (long)64*C_);
  cvt(q_a1, qa1_b, (long)C_*64); cvt(q_a2, qa2_b, (long)64*C_);
  cvt(q_g1, qg1_b, (long)C_*128); cvt(q_g2, qg2_b, (long)128*C_);
  cvt(ffn_Wk, fWk_b, (long)C_*FF_); cvt(ffn_Wv, fWv_b, (long)FF_*C_);

  const float* conv_tok = kvx + (long)(TKV_-1)*C_;   // shifted = kv_x[:, -1]
  const long conv_stride = (long)TKV_ * C_;
  // r
  mix_to_bf16<<<GS(R_Q*C_), 256, 0, stream>>>(qx, q_xmix, 0, conv_tok, conv_stride, mix_q, TQ_, C_, R_Q*C_);
  gemm(mix_q, qWr_b, r_q, (int)R_Q, C_, C_);
  // w lora
  mix_to_bf16<<<GS(R_Q*C_), 256, 0, stream>>>(qx, q_xmix, 1, conv_tok, conv_stride, mix_q, TQ_, C_, R_Q*C_);
  gemm(mix_q, qw1_b, tmp64, (int)R_Q, 64, C_);
  k_tanh_bf16<<<GS(R_Q*64), 256, 0, stream>>>(tmp64, tmp64b, R_Q*64);
  gemm(tmp64b, qw2_b, tmpC_q, (int)R_Q, C_, 64);
  k_decay<<<GS(R_Q*C_), 256, 0, stream>>>(tmpC_q, q_w0, w_q, R_Q*C_, C_);
  // k
  mix_to_bf16<<<GS(R_Q*C_), 256, 0, stream>>>(qx, q_xmix, 2, conv_tok, conv_stride, mix_q, TQ_, C_, R_Q*C_);
  gemm(mix_q, qWk_b, k_q, (int)R_Q, C_, C_);
  // v (query_v_first -> d_out region)
  mix_to_bf16<<<GS(R_Q*C_), 256, 0, stream>>>(qx, q_xmix, 3, conv_tok, conv_stride, mix_q, TQ_, C_, R_Q*C_);
  gemm(mix_q, qWv_b, v_q, (int)R_Q, C_, C_);
  // a lora
  mix_to_bf16<<<GS(R_Q*C_), 256, 0, stream>>>(qx, q_xmix, 4, conv_tok, conv_stride, mix_q, TQ_, C_, R_Q*C_);
  gemm(mix_q, qa1_b, tmp64, (int)R_Q, 64, C_);
  k_cvt_bf16<<<GS(R_Q*64), 256, 0, stream>>>(tmp64, tmp64b, R_Q*64);
  gemm(tmp64b, qa2_b, tmpC_q, (int)R_Q, C_, 64);
  k_sigmoid_bias<<<GS(R_Q*C_), 256, 0, stream>>>(tmpC_q, q_a0, a_q, R_Q*C_, C_);
  // g gate: sigmoid(xg@g1)@g2
  mix_to_bf16<<<GS(R_Q*C_), 256, 0, stream>>>(qx, q_xmix, 5, conv_tok, conv_stride, mix_q, TQ_, C_, R_Q*C_);
  gemm(mix_q, qg1_b, tmp128, (int)R_Q, 128, C_);
  k_sigmoid_bf16<<<GS(R_Q*128), 256, 0, stream>>>(tmp128, tmp128b, R_Q*128);
  gemm(tmp128b, qg2_b, g_q, (int)R_Q, C_, 128);

  prep_heads<<<(unsigned)(R_Q*H_/8), 256, 0, stream>>>(k_q, a_q, q_kk, q_ka, av_q, bv_q, R_Q);
  query_attn_core<<<(unsigned)(R_Q*H_/8), 256, 0, stream>>>(Sst, r_q, w_q, k_q, av_q, bv_q, v_q, ohd, TQ_);
  finish_gate<<<(unsigned)(R_Q*H_/8), 256, 0, stream>>>(ohd, r_q, k_q, v_q, g_q, q_rk, q_gnw, q_gnb, preout);
  gemm(preout, qWo_b, attn, (int)R_Q, C_, C_);
  k_add<<<GS(R_Q*C_), 256, 0, stream>>>(attn, resid0, resid0, R_Q*C_);   // residual = q_attn + residual

  // ============================ FFN ============================
  ln_rows<<<(unsigned)R_Q, 256, 0, stream>>>(resid0, ffnn_w, ffnn_b, hln, C_);
  ffn_mix<<<GS(R_Q*C_), 256, 0, stream>>>(hln, ffn_xk, keyin, TQ_, C_, R_Q*C_);
  gemm(keyin, fWk_b, ffnmid, (int)R_Q, FF_, C_);
  k_sqrelu_bf16<<<GS(R_Q*(long)FF_), 256, 0, stream>>>(ffnmid, ffnmid_b, R_Q*(long)FF_);
  gemm(ffnmid_b, fWv_b, ffnout, (int)R_Q, C_, FF_);
  k_add<<<GS(R_Q*C_), 256, 0, stream>>>(resid0, ffnout, outF, R_Q*C_);
}
